// GATModule_10273561772508
// MI455X (gfx1250) — compile-verified, matrix operand order checked by reading, not run on hardware
//
#include <hip/hip_runtime.h>
#include <math.h>

#define NDIM   128
#define NHEADS 8
#define HIDDIM 512
#define LDSPAD 40   // halfs per panel row (32 data + 8 pad) -> 80B, 16B aligned

typedef __attribute__((ext_vector_type(16))) __bf16 v16bf;
typedef __attribute__((ext_vector_type(8)))  __bf16 v8bf;
typedef __attribute__((ext_vector_type(8)))  float  v8f;

// ---------------- WMMA fragment loaders ------------------------------------
// 16-bit A-matrix 16x32 layout (ISA 7.12.2): m = lane&15, hi = lane>>4;
// halfs[0..7] = K in [hi*8, hi*8+8), halfs[8..15] = K in [hi*8+16, hi*8+24).

__device__ __forceinline__ v16bf load_tile_bf16(const __bf16* base, int ld,
                                                int r0, int k0, int lane) {
  int m = lane & 15, hi = lane >> 4;
  const __bf16* p = base + (size_t)(r0 + m) * ld + k0 + hi * 8;
  v16bf a;
  ((v8bf*)&a)[0] = *(const v8bf*)p;
  ((v8bf*)&a)[1] = *(const v8bf*)(p + 16);
  return a;
}

__device__ __forceinline__ v16bf load_tile_f32(const float* base, int ld,
                                               int r0, int k0, int lane) {
  int m = lane & 15, hi = lane >> 4;
  const float* p = base + (size_t)(r0 + m) * ld + k0 + hi * 8;
  v16bf a;
#pragma unroll
  for (int i = 0; i < 8; ++i) ((__bf16*)&a)[i]     = (__bf16)p[i];
#pragma unroll
  for (int i = 0; i < 8; ++i) ((__bf16*)&a)[8 + i] = (__bf16)p[16 + i];
  return a;
}

// B fragment from padded LDS panel (panel row = output column, 40-half stride)
__device__ __forceinline__ v16bf lds_tile(const __bf16* pan, int col, int lane) {
  int n = lane & 15, hi = lane >> 4;
  const __bf16* p = pan + (size_t)(col + n) * LDSPAD + hi * 8;
  v16bf b;
  ((v8bf*)&b)[0] = *(const v8bf*)p;
  ((v8bf*)&b)[1] = *(const v8bf*)(p + 16);
  return b;
}

__device__ __forceinline__ void wait_async0() {
  asm volatile("s_wait_asynccnt 0x0" ::: "memory");
}

// async-stage one K=32 panel of B columns into LDS (ASYNCcnt tracked)
template <int COLT_BLK>
__device__ __forceinline__ void stage_panel(__bf16* smem, int buf,
                                            const __bf16* BtBlk, int K, int k0) {
  const int PANEL_HALFS = COLT_BLK * 64 * LDSPAD;
  int t = threadIdx.x;
#pragma unroll
  for (int i = 0; i < COLT_BLK; ++i) {           // COLT_BLK 16B chunks / thread
    int c    = t + i * 256;                      // chunk id
    int row  = c >> 2;                           // output column (B row)
    int part = c & 3;                            // 8-half piece within K=32
    const __bf16* g = BtBlk + (size_t)row * K + k0 + part * 8;
    // low 32 bits of a flat LDS address are the LDS byte offset
    uint32_t lds = (uint32_t)(uintptr_t)(smem + buf * PANEL_HALFS +
                                         row * LDSPAD + part * 8);
    asm volatile("global_load_async_to_lds_b128 %0, %1, off"
                 :: "v"(lds), "v"(g) : "memory");
  }
}

// ---------------- WMMA GEMM: block = 8 waves, wave = 16 rows x 64 cols -----
// B (transposed bf16 weights) double-buffered through LDS via async copies.
template <bool A_BF16, bool OUT_BF16, bool GELU, int COLT_BLK>
__global__ void gemm_wmma(const void* __restrict__ Ain,
                          const __bf16* __restrict__ Bt,   // [Ncol][K] bf16
                          const float* __restrict__ bias,
                          void* __restrict__ Out,
                          int M, int K, int Ncol) {
  const int PANEL_HALFS = COLT_BLK * 64 * LDSPAD;
  __shared__ __bf16 smem[2 * COLT_BLK * 64 * LDSPAD];

  const int tid  = threadIdx.x;
  const int lane = tid & 31;
  const int w    = tid >> 5;
  const int colBlocks = Ncol / (COLT_BLK * 64);
  const int rowBlock  = (int)blockIdx.x / colBlocks;
  const int colBlock  = (int)blockIdx.x % colBlocks;
  const int rt = w / COLT_BLK, ct = w % COLT_BLK;

  int r0 = (rowBlock * (8 / COLT_BLK) + rt) * 16;
  if (r0 + 16 > M) r0 = M - 16;                  // idempotent duplicate tile
  const int c0g = colBlock * COLT_BLK * 64 + ct * 64;
  const __bf16* BtBlk = Bt + (size_t)(colBlock * COLT_BLK * 64) * K;

  const int KT = K >> 5;
  v8f acc[4] = {};

  stage_panel<COLT_BLK>(smem, 0, BtBlk, K, 0);
  v16bf a_cur = A_BF16 ? load_tile_bf16((const __bf16*)Ain, K, r0, 0, lane)
                       : load_tile_f32((const float*)Ain, K, r0, 0, lane);
  wait_async0();
  __syncthreads();

  for (int kt = 0; kt < KT; ++kt) {
    if (kt + 1 < KT)                              // overlap: stage next panel
      stage_panel<COLT_BLK>(smem, (kt + 1) & 1, BtBlk, K, (kt + 1) * 32);
    v16bf a_nxt = a_cur;
    if (kt + 1 < KT)                              // prefetch next A fragment
      a_nxt = A_BF16
          ? load_tile_bf16((const __bf16*)Ain, K, r0, (kt + 1) * 32, lane)
          : load_tile_f32((const float*)Ain, K, r0, (kt + 1) * 32, lane);

    const __bf16* pan = smem + (size_t)(kt & 1) * PANEL_HALFS;
    // hoist all 4 B fragments: 8 ds_load_b128 clause, one wait, 4 WMMAs b2b
    v16bf b0 = lds_tile(pan, ct * 64 + 0,  lane);
    v16bf b1 = lds_tile(pan, ct * 64 + 16, lane);
    v16bf b2 = lds_tile(pan, ct * 64 + 32, lane);
    v16bf b3 = lds_tile(pan, ct * 64 + 48, lane);
    acc[0] = __builtin_amdgcn_wmma_f32_16x16x32_bf16(
        false, a_cur, false, b0, (short)0, acc[0], false, false);
    acc[1] = __builtin_amdgcn_wmma_f32_16x16x32_bf16(
        false, a_cur, false, b1, (short)0, acc[1], false, false);
    acc[2] = __builtin_amdgcn_wmma_f32_16x16x32_bf16(
        false, a_cur, false, b2, (short)0, acc[2], false, false);
    acc[3] = __builtin_amdgcn_wmma_f32_16x16x32_bf16(
        false, a_cur, false, b3, (short)0, acc[3], false, false);
    a_cur = a_nxt;
    if (kt + 1 < KT) { wait_async0(); __syncthreads(); }
  }

  // C/D layout: lane&15 = N, (lane>>4)*8 + vgpr = M
  int n  = lane & 15;
  int mB = (lane >> 4) << 3;
#pragma unroll
  for (int t = 0; t < 4; ++t) {
    int col  = c0g + t * 16 + n;
    float bv = bias[col];
#pragma unroll
    for (int r = 0; r < 8; ++r) {
      int row = r0 + mB + r;
      float v = acc[t][r] + bv;
      if (GELU) v = 0.5f * v * (1.0f + erff(v * 0.70710678118654752f));
      if (OUT_BF16) ((__bf16*)Out)[(size_t)row * Ncol + col] = (__bf16)v;
      else          ((float*)Out)[(size_t)row * Ncol + col]  = v;
    }
  }
}

// ---------------- weight prep: fp32 -> bf16 + transpose --------------------
__global__ void prep_weights(const float* __restrict__ W_in,
                             const float* __restrict__ W1,
                             const float* __restrict__ W2,
                             __bf16* __restrict__ WinT,
                             __bf16* __restrict__ W1T,
                             __bf16* __restrict__ W2T) {
  int t = blockIdx.x * blockDim.x + threadIdx.x;
  if (t < NDIM * NDIM) {              // WinT[n][k] = W_in[k][n]
    int n = t / NDIM, k = t % NDIM;
    WinT[t] = (__bf16)W_in[k * NDIM + n];
  }
  if (t < HIDDIM * NDIM) {            // W1T[n][k] = W1[k][n]
    int n = t / NDIM, k = t % NDIM;
    W1T[t] = (__bf16)W1[k * HIDDIM + n];
  }
  if (t < NDIM * HIDDIM) {            // W2T[n][k] = W2[k][n]
    int n = t / HIDDIM, k = t % HIDDIM;
    W2T[t] = (__bf16)W2[k * NDIM + n];
  }
}

__global__ void init_buffers(float* __restrict__ agg, float* __restrict__ m,
                             float* __restrict__ den, int nAgg, int nMD) {
  int total  = nAgg + 2 * nMD;
  int stride = gridDim.x * blockDim.x;
  for (int i = blockIdx.x * blockDim.x + threadIdx.x; i < total; i += stride) {
    if (i < nAgg)            agg[i] = 0.0f;
    else if (i < nAgg + nMD) m[i - nAgg] = -INFINITY;
    else                     den[i - nAgg - nMD] = 0.0f;
  }
}

// ---------------- attention scores -----------------------------------------
__global__ void scores_kernel(const float* __restrict__ x,
                              const float* __restrict__ W_u,
                              const float* __restrict__ b_u,
                              const float* __restrict__ W_v,
                              float* __restrict__ su, float* __restrict__ sv,
                              int N) {
  int t = blockIdx.x * blockDim.x + threadIdx.x;
  if (t >= N * NHEADS) return;
  int n = t >> 3, h = t & 7;
  const float* xp = x + (size_t)n * NDIM;
  float a = b_u[h], b = 0.0f;
#pragma unroll 8
  for (int k = 0; k < NDIM; ++k) {
    float xv = xp[k];
    a += xv * W_u[k * NHEADS + h];
    b += xv * W_v[k * NHEADS + h];
  }
  su[t] = a;
  sv[t] = b;
}

__device__ __forceinline__ void atomicMaxFloat(float* addr, float v) {
  if (v >= 0.0f) atomicMax((int*)addr, __float_as_int(v));
  else           atomicMin((unsigned int*)addr, __float_as_uint(v));
}

__global__ void edge_pass1(const int* __restrict__ src, const int* __restrict__ dst,
                           const float* __restrict__ su, const float* __restrict__ sv,
                           float* __restrict__ ebuf, float* __restrict__ mbuf, int E) {
  int e = blockIdx.x * blockDim.x + threadIdx.x;
  if (e >= E) return;
  int s = src[e], d = dst[e];
  const float* sp = su + (size_t)s * 8;
  const float* vp = sv + (size_t)d * 8;
  float* ep = ebuf + (size_t)e * 8;
  float* mp = mbuf + (size_t)d * 8;
#pragma unroll
  for (int h = 0; h < 8; ++h) {
    float v = sp[h] + vp[h];
    v = v > 0.0f ? v : 0.2f * v;
    ep[h] = v;
    atomicMaxFloat(mp + h, v);
  }
}

__global__ void edge_pass2(const int* __restrict__ dst, float* __restrict__ ebuf,
                           const float* __restrict__ mbuf, float* __restrict__ den,
                           int E) {
  int e = blockIdx.x * blockDim.x + threadIdx.x;
  if (e >= E) return;
  int d = dst[e];
  float* ep = ebuf + (size_t)e * 8;
  const float* mp = mbuf + (size_t)d * 8;
  float* dp = den + (size_t)d * 8;
#pragma unroll
  for (int h = 0; h < 8; ++h) {
    float ex = expf(ep[h] - mp[h]);
    ep[h] = ex;
    atomicAdd(dp + h, ex);
  }
}

// wave per edge; lane handles features [4*lane, 4*lane+4); head = d&7
__global__ void edge_aggregate(const int* __restrict__ src, const int* __restrict__ dst,
                               const float* __restrict__ x, const float* __restrict__ ebuf,
                               const float* __restrict__ den, float* __restrict__ agg,
                               int E) {
  int gid  = blockIdx.x * blockDim.x + threadIdx.x;
  int e    = gid >> 5;
  int lane = gid & 31;
  if (e >= E) return;
  int s = src[e], d = dst[e];
  int hb = (lane & 1) << 2;
  const float* exq = ebuf + (size_t)e * 8 + hb;
  const float* dq  = den  + (size_t)d * 8 + hb;
  float p0 = exq[0] / dq[0], p1 = exq[1] / dq[1];
  float p2 = exq[2] / dq[2], p3 = exq[3] / dq[3];
  int d0 = lane << 2;
  const float* xp = x   + (size_t)s * NDIM + d0;
  float*       ap = agg + (size_t)d * NDIM + d0;
  atomicAdd(ap + 0, xp[0] * p0);
  atomicAdd(ap + 1, xp[1] * p1);
  atomicAdd(ap + 2, xp[2] * p2);
  atomicAdd(ap + 3, xp[3] * p3);
}

// ---------------------------------------------------------------------------
extern "C" void kernel_launch(void* const* d_in, const int* in_sizes, int n_in,
                              void* d_out, int out_size, void* d_ws, size_t ws_size,
                              hipStream_t stream) {
  const float* h    = (const float*)d_in[0];
  const int*   src  = (const int*)d_in[1];
  const int*   dst  = (const int*)d_in[2];
  const float* W_in = (const float*)d_in[3];
  const float* b_in = (const float*)d_in[4];
  const float* W_u  = (const float*)d_in[5];
  const float* b_u  = (const float*)d_in[6];
  const float* W_v  = (const float*)d_in[7];
  const float* W1   = (const float*)d_in[8];
  const float* b1   = (const float*)d_in[9];
  const float* W2   = (const float*)d_in[10];
  const float* b2   = (const float*)d_in[11];
  float* y = (float*)d_out;

  const int N = in_sizes[0] / NDIM;   // 100000
  const int E = in_sizes[1];          // 1600000

  char* ws = (char*)d_ws;
  size_t offX   = 0;
  size_t offE   = offX   + (size_t)N * NDIM * 4;
  size_t offAgg = offE   + (size_t)E * 8 * 4;
  size_t offSu  = offAgg + (size_t)N * NDIM * 4;
  size_t offSv  = offSu  + (size_t)N * 8 * 4;
  size_t offM   = offSv  + (size_t)N * 8 * 4;
  size_t offD   = offM   + (size_t)N * 8 * 4;
  size_t offWin = offD   + (size_t)N * 8 * 4;
  size_t offW1  = offWin + (size_t)NDIM * NDIM * 2;
  size_t offW2  = offW1  + (size_t)HIDDIM * NDIM * 2;

  float*  x    = (float*)(ws + offX);
  float*  ebuf = (float*)(ws + offE);
  float*  agg  = (float*)(ws + offAgg);
  float*  su   = (float*)(ws + offSu);
  float*  sv   = (float*)(ws + offSv);
  float*  mbuf = (float*)(ws + offM);
  float*  den  = (float*)(ws + offD);
  __bf16* WinT = (__bf16*)(ws + offWin);
  __bf16* W1T  = (__bf16*)(ws + offW1);
  __bf16* W2T  = (__bf16*)(ws + offW2);
  __bf16* hid  = (__bf16*)(ws + offX);   // aliases x+ebuf (dead by then)

  prep_weights<<<(HIDDIM * NDIM + 255) / 256, 256, 0, stream>>>(W_in, W1, W2,
                                                                WinT, W1T, W2T);
  init_buffers<<<8192, 256, 0, stream>>>(agg, mbuf, den, N * NDIM, N * NHEADS);

  const int rowTiles = N / 16;  // 6250

  // 1) x = h @ W_in + b_in : COLT_BLK=2 -> 4 row tiles/block, 1 col block
  {
    int rowBlocks = (rowTiles + 3) / 4;
    gemm_wmma<false, false, false, 2><<<rowBlocks, 256, 0, stream>>>(
        h, WinT, b_in, x, N, NDIM, NDIM);
  }

  scores_kernel<<<(N * NHEADS + 255) / 256, 256, 0, stream>>>(x, W_u, b_u, W_v,
                                                              su, sv, N);

  edge_pass1<<<(E + 255) / 256, 256, 0, stream>>>(src, dst, su, sv, ebuf, mbuf, E);
  edge_pass2<<<(E + 255) / 256, 256, 0, stream>>>(dst, ebuf, mbuf, den, E);
  edge_aggregate<<<((size_t)E * 32 + 255) / 256, 256, 0, stream>>>(src, dst, x,
                                                                   ebuf, den, agg, E);

  // 4) hid = gelu(agg @ W1 + b1) bf16 : COLT_BLK=4 -> 2 row tiles/block, 2 col blocks
  {
    int rowBlocks = (rowTiles + 1) / 2;
    gemm_wmma<false, true, true, 4><<<rowBlocks * 2, 256, 0, stream>>>(
        agg, W1T, b1, hid, N, NDIM, HIDDIM);
  }
  // 5) y = hid @ W2 + b2 : COLT_BLK=2
  {
    int rowBlocks = (rowTiles + 3) / 4;
    gemm_wmma<true, false, false, 2><<<rowBlocks, 256, 0, stream>>>(
        hid, W2T, b2, y, N, HIDDIM, NDIM);
  }
}
